// WindSpeedLSTM_30683246363144
// MI455X (gfx1250) — matrix operand order repaired and checked by reading, not compiled, not run on hardware
//
#include <hip/hip_runtime.h>
#include <hip/hip_bf16.h>
#include <math.h>
#include <stdint.h>

#define H      512
#define BATCH  16
#define NWG    32
#define TPB    128   // 4 wave32 waves: one per gate (i,f,g,o)
#define HC     16    // h-columns owned per workgroup per layer

typedef unsigned short bfbits;  // raw bf16 bits (storage)
typedef __attribute__((ext_vector_type(16))) __bf16 v16bf;
typedef __attribute__((ext_vector_type(8)))  __bf16 v8bf;
typedef __attribute__((ext_vector_type(8)))  float  v8f;

union Frag { v16bf v; v8bf h[2]; };

// ---- LDS layout (bytes) ----
// Wb    :      0 .. 196608   3 matrices * 4 gates * 16 ksteps * 512 bf16 (pre-swizzled B frags)
// hA1   : 196608 .. 212992   16x512 bf16 A-staging
// hA2   : 212992 .. 229376   16x512 bf16 A-staging
// Dg    : 229376 .. 233472   4 * 256 f32 gate tiles
// c1s   : 233472 .. 234496   256 f32 cell state L1
// c2s   : 234496 .. 235520   256 f32 cell state L2
// b1s   : 235520 .. 235776   64 f32 combined bias L1
// b2s   : 235776 .. 236032   64 f32 combined bias L2
// wi1   : 236032 .. 236544   64*2 f32 W_ih1 slice
// wouts : 236544 .. 236608   16 f32 W_out slice
// xs    : 236608 .. 236736   32 f32 x_t
// htile : 236736 .. 237760   256 f32 h2 tile for output dot
#define SMEM_BYTES 237760

__device__ __forceinline__ bfbits f2bf(float f) {
  unsigned u = __builtin_bit_cast(unsigned, f);
  unsigned r = (u + 0x7FFFu + ((u >> 16) & 1u)) >> 16;   // RNE
  return (bfbits)r;
}

__device__ __forceinline__ float sigf(float x) { return 1.f / (1.f + __expf(-x)); }

__device__ __forceinline__ void grid_sync(unsigned* cnt, unsigned target) {
  __syncthreads();
  __threadfence();                       // make this block's global h-stores visible
  if (threadIdx.x == 0) {
    __hip_atomic_fetch_add(cnt, 1u, __ATOMIC_ACQ_REL, __HIP_MEMORY_SCOPE_AGENT);
    while (__hip_atomic_load(cnt, __ATOMIC_ACQUIRE, __HIP_MEMORY_SCOPE_AGENT) < target) {
      __builtin_amdgcn_s_sleep(2);
    }
  }
  __syncthreads();
}

// One 16x16x32 bf16 WMMA K-step: A from staged h (hA), B from swizzled LDS weights.
__device__ __forceinline__ v8f wmma_k(const bfbits* __restrict__ hA,
                                      const bfbits* __restrict__ Wt,
                                      int ks, int M, int hi, int lane, v8f acc) {
  Frag A, B;
  const bfbits* ap = hA + M * H + ks * 32 + hi * 16;
  A.h[0] = *(const v8bf*)ap;
  A.h[1] = *(const v8bf*)(ap + 8);
  const bfbits* bp = Wt + ks * 512 + lane * 16;
  B.h[0] = *(const v8bf*)bp;
  B.h[1] = *(const v8bf*)(bp + 8);
  return __builtin_amdgcn_wmma_f32_16x16x32_bf16(false, A.v, false, B.v,
                                                 (short)0, acc, false, false);
}

__global__ void __launch_bounds__(TPB, 1)
init_kernel(bfbits* h1g, bfbits* h2g, unsigned* cnt, float* out,
            const float* b_out, int total_out) {
  int i = blockIdx.x * blockDim.x + threadIdx.x;
  int stride = gridDim.x * blockDim.x;
  for (int j = i; j < 2 * BATCH * H; j += stride) { h1g[j] = 0; h2g[j] = 0; }
  float bo = b_out[0];
  for (int j = i; j < total_out; j += stride) out[j] = bo;
  if (i == 0) *cnt = 0u;
}

__global__ void __launch_bounds__(TPB, 1)
lstm_persistent(const float* __restrict__ input,
                const float* __restrict__ W_ih1, const float* __restrict__ W_hh1,
                const float* __restrict__ b_ih1, const float* __restrict__ b_hh1,
                const float* __restrict__ W_ih2, const float* __restrict__ W_hh2,
                const float* __restrict__ b_ih2, const float* __restrict__ b_hh2,
                const float* __restrict__ W_out,
                float* __restrict__ out,
                bfbits* __restrict__ h1g,   // [2][16][512] double buffered
                bfbits* __restrict__ h2g,   // [2][16][512]
                unsigned* __restrict__ cnt,
                int T) {
  extern __shared__ char smem[];
  bfbits* Wb    = (bfbits*)(smem);
  bfbits* hA1   = (bfbits*)(smem + 196608);
  bfbits* hA2   = (bfbits*)(smem + 212992);
  float*  Dg    = (float*)(smem + 229376);
  float*  c1s   = (float*)(smem + 233472);
  float*  c2s   = (float*)(smem + 234496);
  float*  b1s   = (float*)(smem + 235520);
  float*  b2s   = (float*)(smem + 235776);
  float*  wi1   = (float*)(smem + 236032);
  float*  wouts = (float*)(smem + 236544);
  float*  xs    = (float*)(smem + 236608);
  float*  htile = (float*)(smem + 236736);

  const int tid  = threadIdx.x;
  const int lane = tid & 31;
  const int wv   = tid >> 5;        // gate 0..3 (i,f,g,o)
  const int wg   = blockIdx.x;      // 0..31 -> h-columns [wg*16, wg*16+16)

  // ---------- one-time: swizzle weight slices into LDS as WMMA B-fragments ----------
  {
    const float* Wsrc[3] = { W_hh1, W_ih2, W_hh2 };
    for (int i = tid; i < 3 * 4 * 16 * 512; i += TPB) {
      int m  = i >> 15;         // matrix
      int r  = i & 32767;
      int g  = r >> 13;         // gate
      int r2 = r & 8191;
      int ks = r2 >> 9;         // k-step (32 K per step)
      int q  = r2 & 511;
      int ln = q >> 4;          // target lane
      int e  = q & 15;          // element within lane
      int row = g * 512 + wg * HC + (ln & 15);          // N (gate column) = W row
      int col = ks * 32 + (ln >> 4) * 16 + e;           // K
      Wb[i] = f2bf(Wsrc[m][row * 512 + col]);
    }
    for (int i = tid; i < 4 * HC; i += TPB) {
      int g = i >> 4, n = i & 15;
      int row = g * 512 + wg * HC + n;
      b1s[i] = b_ih1[row] + b_hh1[row];
      b2s[i] = b_ih2[row] + b_hh2[row];
      wi1[i * 2 + 0] = W_ih1[row * 2 + 0];
      wi1[i * 2 + 1] = W_ih1[row * 2 + 1];
    }
    if (tid < HC) wouts[tid] = W_out[wg * HC + tid];
    for (int i = tid; i < 256; i += TPB) { c1s[i] = 0.f; c2s[i] = 0.f; }
  }
  __syncthreads();

  const int M  = lane & 15;      // A-fragment row (batch index)
  const int hi = lane >> 4;      // lane half selector

  unsigned target = 0;
  for (int t = 0; t < T; ++t) {
    const bfbits* h1prev = h1g + (t & 1) * (BATCH * H);
    bfbits*       h1next = h1g + ((t + 1) & 1) * (BATCH * H);
    const bfbits* h2prev = h2g + (t & 1) * (BATCH * H);
    bfbits*       h2next = h2g + ((t + 1) & 1) * (BATCH * H);

    // ---------- stage x_t and h1(t-1) ----------
    if (tid < 32) {
      int b = tid >> 1, c = tid & 1;
      xs[tid] = input[(b * T + t) * 2 + c];
    }
    {
      const uint4* s = (const uint4*)h1prev;
      uint4* d = (uint4*)hA1;
      for (int i = tid; i < BATCH * H / 8; i += TPB) d[i] = s[i];
    }
    __syncthreads();

    // ---------- layer 1: gate tile (16 batch x 16 cols) = h1 @ W_hh1^T slice ----------
    // Two independent accumulator chains so consecutive WMMAs have no D->C RAW hazard.
    {
      v8f acc0 = {}, acc1 = {};
      const bfbits* Wt = Wb + ((0 * 4 + wv) * 16) * 512;
      #pragma unroll 8
      for (int ks = 0; ks < 16; ks += 2) {
        acc0 = wmma_k(hA1, Wt, ks + 0, M, hi, lane, acc0);
        acc1 = wmma_k(hA1, Wt, ks + 1, M, hi, lane, acc1);
      }
      v8f acc = acc0 + acc1;
      int n = lane & 15, mh = hi * 8;
      #pragma unroll
      for (int v = 0; v < 8; ++v) Dg[wv * 256 + (v + mh) * 16 + n] = acc[v];
    }
    __syncthreads();

    // ---------- layer 1 elementwise: add bias + x@W_ih1^T, LSTM cell ----------
    for (int idx = tid; idx < 256; idx += TPB) {
      int b = idx >> 4, n = idx & 15;
      float x0 = xs[b * 2 + 0], x1 = xs[b * 2 + 1];
      float gi = Dg[0 * 256 + idx] + b1s[0 * 16 + n] + x0 * wi1[(0 * 16 + n) * 2] + x1 * wi1[(0 * 16 + n) * 2 + 1];
      float gf = Dg[1 * 256 + idx] + b1s[1 * 16 + n] + x0 * wi1[(1 * 16 + n) * 2] + x1 * wi1[(1 * 16 + n) * 2 + 1];
      float gg = Dg[2 * 256 + idx] + b1s[2 * 16 + n] + x0 * wi1[(2 * 16 + n) * 2] + x1 * wi1[(2 * 16 + n) * 2 + 1];
      float go = Dg[3 * 256 + idx] + b1s[3 * 16 + n] + x0 * wi1[(3 * 16 + n) * 2] + x1 * wi1[(3 * 16 + n) * 2 + 1];
      float iv = sigf(gi), fv = sigf(gf), gv = tanhf(gg), ov = sigf(go);
      float c = fv * c1s[idx] + iv * gv;
      c1s[idx] = c;
      float h = ov * tanhf(c);
      h1next[b * H + wg * HC + n] = f2bf(h);
    }

    target += NWG;
    grid_sync(cnt, target);          // h1(t) now globally visible

    // ---------- stage h1(t) and h2(t-1) ----------
    {
      const uint4* s1 = (const uint4*)h1next;
      uint4* d1 = (uint4*)hA1;
      const uint4* s2 = (const uint4*)h2prev;
      uint4* d2 = (uint4*)hA2;
      for (int i = tid; i < BATCH * H / 8; i += TPB) { d1[i] = s1[i]; d2[i] = s2[i]; }
    }
    __syncthreads();

    // ---------- layer 2: h1(t) @ W_ih2^T + h2(t-1) @ W_hh2^T ----------
    {
      v8f acc0 = {}, acc1 = {};
      const bfbits* Wt1 = Wb + ((1 * 4 + wv) * 16) * 512;
      const bfbits* Wt2 = Wb + ((2 * 4 + wv) * 16) * 512;
      #pragma unroll 8
      for (int ks = 0; ks < 16; ks += 2) {
        acc0 = wmma_k(hA1, Wt1, ks + 0, M, hi, lane, acc0);
        acc1 = wmma_k(hA1, Wt1, ks + 1, M, hi, lane, acc1);
      }
      #pragma unroll 8
      for (int ks = 0; ks < 16; ks += 2) {
        acc0 = wmma_k(hA2, Wt2, ks + 0, M, hi, lane, acc0);
        acc1 = wmma_k(hA2, Wt2, ks + 1, M, hi, lane, acc1);
      }
      v8f acc = acc0 + acc1;
      int n = lane & 15, mh = hi * 8;
      #pragma unroll
      for (int v = 0; v < 8; ++v) Dg[wv * 256 + (v + mh) * 16 + n] = acc[v];
    }
    __syncthreads();

    // ---------- layer 2 elementwise + h2 write ----------
    for (int idx = tid; idx < 256; idx += TPB) {
      int b = idx >> 4, n = idx & 15;
      float gi = Dg[0 * 256 + idx] + b2s[0 * 16 + n];
      float gf = Dg[1 * 256 + idx] + b2s[1 * 16 + n];
      float gg = Dg[2 * 256 + idx] + b2s[2 * 16 + n];
      float go = Dg[3 * 256 + idx] + b2s[3 * 16 + n];
      float iv = sigf(gi), fv = sigf(gf), gv = tanhf(gg), ov = sigf(go);
      float c = fv * c2s[idx] + iv * gv;
      c2s[idx] = c;
      float h = ov * tanhf(c);
      htile[idx] = h;
      h2next[b * H + wg * HC + n] = f2bf(h);
    }
    __syncthreads();

    // ---------- partial output: this WG's 16-col contribution to h2 . w_out ----------
    if (tid < BATCH) {
      float s = 0.f;
      #pragma unroll
      for (int n = 0; n < HC; ++n) s += htile[tid * HC + n] * wouts[n];
      atomicAdd(&out[tid * T + t], s);
    }

    target += NWG;
    grid_sync(cnt, target);          // h2(t) globally visible before next step's layer 2
  }
}

extern "C" void kernel_launch(void* const* d_in, const int* in_sizes, int n_in,
                              void* d_out, int out_size, void* d_ws, size_t ws_size,
                              hipStream_t stream) {
  const float* input = (const float*)d_in[0];
  const float* W_ih1 = (const float*)d_in[1];
  const float* W_hh1 = (const float*)d_in[2];
  const float* b_ih1 = (const float*)d_in[3];
  const float* b_hh1 = (const float*)d_in[4];
  const float* W_ih2 = (const float*)d_in[5];
  const float* W_hh2 = (const float*)d_in[6];
  const float* b_ih2 = (const float*)d_in[7];
  const float* b_hh2 = (const float*)d_in[8];
  const float* W_out = (const float*)d_in[9];
  const float* b_out = (const float*)d_in[10];
  float* out = (float*)d_out;

  const int T = in_sizes[0] / (BATCH * 2);

  bfbits*   h1g = (bfbits*)d_ws;                                   // 2*16*512 bf16
  bfbits*   h2g = h1g + 2 * BATCH * H;                             // 2*16*512 bf16
  unsigned* cnt = (unsigned*)((char*)d_ws + 4 * BATCH * H * sizeof(bfbits));

  init_kernel<<<64, TPB, 0, stream>>>(h1g, h2g, cnt, out, b_out, out_size);

  lstm_persistent<<<NWG, TPB, SMEM_BYTES, stream>>>(
      input, W_ih1, W_hh1, b_ih1, b_hh1, W_ih2, W_hh2, b_ih2, b_hh2,
      W_out, out, h1g, h2g, cnt, T);
}
